// Attention_54142357733578
// MI455X (gfx1250) — matrix operand order repaired and checked by reading, not compiled, not run on hardware
//
#include <hip/hip_runtime.h>

// Problem constants (match reference)
#define BATCH 4
#define SEQ   2048
#define DIM   1024
#define DKK   256   // DIM / 4

typedef __attribute__((ext_vector_type(16))) __bf16 v16bf;
typedef __attribute__((ext_vector_type(8)))  float  v8f;
typedef __attribute__((ext_vector_type(4)))  unsigned int u32x4;
typedef __attribute__((ext_vector_type(8)))  int          i32x8;
typedef __attribute__((ext_vector_type(4)))  int          i32x4;

// ---------------------------------------------------------------------------
// CDNA5 WMMA fragment helpers (wave32 layouts per CDNA5 ISA)
// ---------------------------------------------------------------------------

// A-matrix 16x32 bf16, row-major source, leading dim lda.
// lane r=lane&15 holds row M=r; half=lane>>4 selects K octets:
//   elems 0..7  -> K = 8*half + e ; elems 8..15 -> K = 16 + 8*half + (e-8)
// Both octets are contiguous in memory -> 2x 16B vector loads per lane.
static __device__ __forceinline__ v16bf load_a_bf16(const __bf16* A, int lda) {
  const int lane = threadIdx.x & 31;
  const int r = lane & 15;
  const int h = lane >> 4;
  const __bf16* row = A + (size_t)r * lda;
  v16bf a;
#pragma unroll
  for (int e = 0; e < 8; ++e) a[e] = row[8 * h + e];
#pragma unroll
  for (int e = 0; e < 8; ++e) a[8 + e] = row[16 + 8 * h + e];
  return a;
}

// Same A layout but source is fp32 (LDS softmax probs) -> convert to bf16.
static __device__ __forceinline__ v16bf load_a_f32_as_bf16(const float* A, int lda) {
  const int lane = threadIdx.x & 31;
  const int r = lane & 15;
  const int h = lane >> 4;
  const float* row = A + (size_t)r * lda;
  v16bf a;
#pragma unroll
  for (int e = 0; e < 8; ++e) a[e] = (__bf16)row[8 * h + e];
#pragma unroll
  for (int e = 0; e < 8; ++e) a[8 + e] = (__bf16)row[16 + 8 * h + e];
  return a;
}

// B-matrix 32x16 bf16 (KxN) with strides. lane n=lane&15 holds column N=n;
// half=lane>>4: elems 0..7 -> K = 8*half + e ; elems 8..15 -> K = 16 + 8*half + (e-8).
// With sk==1 these are two contiguous 16B runs per lane.
static __device__ __forceinline__ v16bf load_b_bf16(const __bf16* Bp, int sk, int sn) {
  const int lane = threadIdx.x & 31;
  const int n = lane & 15;
  const int h = lane >> 4;
  const __bf16* p = Bp + (size_t)n * sn + (size_t)(8 * h) * sk;
  v16bf b;
#pragma unroll
  for (int e = 0; e < 8; ++e) b[e] = p[(size_t)e * sk];
#pragma unroll
  for (int e = 0; e < 8; ++e) b[8 + e] = p[(size_t)(16 + e) * sk];
  return b;
}

// C/D 16x16 f32: VGPR i, lane (n=lane&15, half=lane>>4) -> M = i + 8*half, N = n.
static __device__ __forceinline__ void store_c_f32(float* C, int ldc, v8f acc) {
  const int lane = threadIdx.x & 31;
  const int n = lane & 15;
  const int h = lane >> 4;
#pragma unroll
  for (int i = 0; i < 8; ++i) C[(size_t)(8 * h + i) * ldc + n] = acc[i];
}

static __device__ __forceinline__ void store_c_bf16(__bf16* C, int ldc, v8f acc) {
  const int lane = threadIdx.x & 31;
  const int n = lane & 15;
  const int h = lane >> 4;
#pragma unroll
  for (int i = 0; i < 8; ++i) C[(size_t)(8 * h + i) * ldc + n] = (__bf16)acc[i];
}

static __device__ __forceinline__ v8f wmma_bf16(v16bf a, v16bf b, v8f c) {
  // (neg_a, A, neg_b, B, c_mod, C, reuse_a, reuse_b)
  return __builtin_amdgcn_wmma_f32_16x16x32_bf16(false, a, false, b, (short)0, c,
                                                 false, false);
}

// ---------------------------------------------------------------------------
// Tensor Data Mover: 2D tile (bf16 elements) global -> LDS.
// D# layout per CDNA5 ISA:
//   g0: [1:0]=count=1 | [63:32]=lds_addr | [120:64]=global_addr | [127:126]=type=2
//   g1: [17:16]=data_size=1 (2B) | [79:48]=tensor_dim0 | [111:80]=tensor_dim1 |
//       [127:112]=tile_dim0 | [143:128]=tile_dim1 | [207:160]=tensor_dim0_stride
// This toolchain exposes the 6-arg builtin:
//   (u32x4 g0, i32x8 g1, i32x4 g2, i32x4 g3, i32x8 extra, i32 cpol)
// ---------------------------------------------------------------------------
#if __has_builtin(__builtin_amdgcn_tensor_load_to_lds)
#define HAVE_TDM 1
static __device__ __forceinline__ void tdm_load_2d_bf16(unsigned lds_byte_off,
                                                        const void* gaddr,
                                                        unsigned rows,
                                                        unsigned row_elems,
                                                        unsigned row_stride_elems) {
  const unsigned long long ga = (unsigned long long)gaddr;
  u32x4 g0;
  g0[0] = 1u;                                                   // count=1
  g0[1] = lds_byte_off;                                         // lds_addr
  g0[2] = (unsigned)(ga & 0xFFFFFFFFu);                         // gaddr[31:0]
  g0[3] = (unsigned)((ga >> 32) & 0x01FFFFFFu) | (2u << 30);    // gaddr[56:32]|type=2
  i32x8 g1;
  g1[0] = (int)(1u << 16);                                      // data_size=1 (2B)
  g1[1] = (int)((row_elems & 0xFFFFu) << 16);                   // tensor_dim0 lo
  g1[2] = (int)(((row_elems >> 16) & 0xFFFFu) |                 // tensor_dim0 hi
                ((rows & 0xFFFFu) << 16));                      // tensor_dim1 lo
  g1[3] = (int)(((rows >> 16) & 0xFFFFu) |                      // tensor_dim1 hi
                ((row_elems & 0xFFFFu) << 16));                 // tile_dim0
  g1[4] = (int)(rows & 0xFFFFu);                                // tile_dim1, tile_dim2=0
  g1[5] = (int)row_stride_elems;                                // dim0_stride[31:0]
  g1[6] = 0;                                                    // dim0_stride[47:32]
  g1[7] = 0;
  i32x4 z4 = {0, 0, 0, 0};
  i32x8 z8 = {0, 0, 0, 0, 0, 0, 0, 0};
  __builtin_amdgcn_tensor_load_to_lds(g0, g1, z4, z4, z8, 0);
}
#else
#define HAVE_TDM 0
#endif

// ---------------------------------------------------------------------------
// 1) fp32 -> bf16 conversion
// ---------------------------------------------------------------------------
__global__ void cvt_f32_to_bf16(const float* __restrict__ src,
                                __bf16* __restrict__ dst, int n) {
  int i = blockIdx.x * blockDim.x + threadIdx.x;
  if (i < n) dst[i] = (__bf16)src[i];
}

// ---------------------------------------------------------------------------
// 1b) Fused convert + transpose: q[b][s][d] f32 -> Vt[b][d][s] bf16.
//     32x32 tiles through LDS, both sides coalesced. block = (32,8).
// ---------------------------------------------------------------------------
__global__ void transpose_cvt_bf16(const float* __restrict__ q,
                                   __bf16* __restrict__ Vt) {
  __shared__ float tile[32][33];
  const int b = blockIdx.z;
  const int s0 = blockIdx.x * 32;
  const int d0 = blockIdx.y * 32;
#pragma unroll
  for (int i = threadIdx.y; i < 32; i += 8)
    tile[i][threadIdx.x] =
        q[((size_t)b * SEQ + s0 + i) * DIM + d0 + threadIdx.x];
  __syncthreads();
#pragma unroll
  for (int i = threadIdx.y; i < 32; i += 8)
    Vt[((size_t)b * DIM + d0 + i) * SEQ + s0 + threadIdx.x] =
        (__bf16)tile[threadIdx.x][i];
}

// ---------------------------------------------------------------------------
// 2) C[M,N] = A[M,K](bf16,row-major) * W[N,K]^T (bf16, nn.Linear layout)
//    block = 128 threads = 4 waves; wave computes a 16x64 tile.
//    The 16xK A tile (32 KB, K==DIM) is staged to LDS once per workgroup by
//    the Tensor Data Mover; waves then read A fragments with ds_load_b128.
//    Dynamic LDS only -> staged tile lives at LDS offset 0 (== D#.lds_addr).
// ---------------------------------------------------------------------------
template <bool OUT_BF16>
__global__ void gemm_a_bt(const __bf16* __restrict__ A,
                          const __bf16* __restrict__ W,
                          void* __restrict__ C, int M, int N, int K) {
  extern __shared__ __bf16 sA[];  // 16 x K
  const int wave = threadIdx.x >> 5;
  const int m0 = blockIdx.y * 16;
  const int n0 = blockIdx.x * 256 + wave * 64;

#if HAVE_TDM
  if (threadIdx.x < 32) {  // one TDM request per workgroup (EXEC ignored by TDM)
    tdm_load_2d_bf16(0u, A + (size_t)m0 * K, 16u, (unsigned)K, (unsigned)K);
    __builtin_amdgcn_s_wait_tensorcnt(0);
  }
#else
  {  // cooperative fallback: rows m0..m0+15 are one contiguous 32KB region
    const uint4* src = (const uint4*)(A + (size_t)m0 * K);
    uint4* dst = (uint4*)sA;
    const int n16 = 16 * K * 2 / 16;
    for (int i = threadIdx.x; i < n16; i += blockDim.x) dst[i] = src[i];
  }
#endif
  __syncthreads();

  v8f acc[4] = {};
  for (int k0 = 0; k0 < K; k0 += 32) {
    v16bf a = load_a_bf16(sA + k0, K);
    __builtin_prefetch(W + (size_t)n0 * K + k0 + 256, 0, 0);  // global_prefetch
#pragma unroll
    for (int t = 0; t < 4; ++t) {
      // B[k][n] = W[n0+16t+n][k0+k] : strideK = 1 (contiguous), strideN = K
      v16bf b = load_b_bf16(W + (size_t)(n0 + 16 * t) * K + k0, 1, K);
      acc[t] = wmma_bf16(a, b, acc[t]);
    }
  }
#pragma unroll
  for (int t = 0; t < 4; ++t) {
    if constexpr (OUT_BF16)
      store_c_bf16((__bf16*)C + (size_t)m0 * N + n0 + 16 * t, N, acc[t]);
    else
      store_c_f32((float*)C + (size_t)m0 * N + n0 + 16 * t, N, acc[t]);
  }
}

// ---------------------------------------------------------------------------
// 3) Fused attention for one (batch, 16-query-row) tile. 256 threads = 8 waves.
//    Phase A: wave w -> scores for keys [w*256, w*256+256) into LDS fp32
//             (scaled 1/sqrt(DK)); QP A-fragments hoisted to registers.
//    Phase B: row softmax in LDS, 16 threads/row, 16-wide xor shuffles.
//    Phase C: O[16x1024] = P @ V with V^T layout: wave owns 128 columns,
//             B fragments are contiguous runs of Vt -> global_load_b128.
//    Dynamic LDS: 16*SEQ*4 = 128 KB of the 320 KB WGP pool.
// ---------------------------------------------------------------------------
__global__ void attn_softmax_pv(const __bf16* __restrict__ QP,
                                const __bf16* __restrict__ KP,
                                const __bf16* __restrict__ Vt,
                                __bf16* __restrict__ Obf) {
  extern __shared__ float sS[];  // [16][SEQ]
  const int b = blockIdx.y;
  const int q0 = blockIdx.x * 16;
  const int tid = threadIdx.x;
  const int wave = tid >> 5;

  const __bf16* qp = QP + ((size_t)b * SEQ + q0) * DKK;
  const __bf16* kp = KP + (size_t)b * SEQ * DKK;
  const __bf16* vt = Vt + (size_t)b * DIM * SEQ;

  // ---- Phase A: scores ----
  v16bf afrag[DKK / 32];
#pragma unroll
  for (int kk = 0; kk < DKK / 32; ++kk)
    afrag[kk] = load_a_bf16(qp + kk * 32, DKK);

  const float scale = 0.0625f;  // 1/sqrt(256)
  const int lane = tid & 31;
  const int cn = lane & 15;
  const int ch = lane >> 4;
  for (int kt = 0; kt < 16; ++kt) {
    const int key0 = wave * 256 + kt * 16;
    v8f acc = {};
#pragma unroll
    for (int kk = 0; kk < DKK / 32; ++kk) {
      // B[k][n] = KP[key0+n][kk*32+k] : strideK=1, strideN=DKK
      v16bf bf = load_b_bf16(kp + (size_t)key0 * DKK + kk * 32, 1, DKK);
      acc = wmma_bf16(afrag[kk], bf, acc);
    }
#pragma unroll
    for (int i = 0; i < 8; ++i)
      sS[(size_t)(8 * ch + i) * SEQ + key0 + cn] = acc[i] * scale;
  }
  __syncthreads();

  // ---- Phase B: softmax (row = tid>>4, 16 threads per row) ----
  {
    const int row = tid >> 4;
    const int sub = tid & 15;
    float* r = sS + (size_t)row * SEQ;
    float mx = -3.402823466e38f;
    for (int c = sub; c < SEQ; c += 16) mx = fmaxf(mx, r[c]);
#pragma unroll
    for (int m = 8; m >= 1; m >>= 1) mx = fmaxf(mx, __shfl_xor(mx, m, 16));
    float sum = 0.0f;
    for (int c = sub; c < SEQ; c += 16) {
      float e = __expf(r[c] - mx);
      r[c] = e;
      sum += e;
    }
#pragma unroll
    for (int m = 8; m >= 1; m >>= 1) sum += __shfl_xor(sum, m, 16);
    const float inv = 1.0f / sum;
    for (int c = sub; c < SEQ; c += 16) r[c] *= inv;
  }
  __syncthreads();

  // ---- Phase C: O = P @ V via V^T (wave owns 128 output columns) ----
  const int n0 = wave * 128;
  v8f acc[8] = {};
  for (int k0 = 0; k0 < SEQ; k0 += 32) {
    v16bf a = load_a_f32_as_bf16(sS + k0, SEQ);
#pragma unroll
    for (int t = 0; t < 8; ++t) {
      // B[k][n] = V[k][n] = Vt[n0+16t+n][k0+k] : strideK=1, strideN=SEQ
      v16bf bf = load_b_bf16(vt + (size_t)(n0 + 16 * t) * SEQ + k0, 1, SEQ);
      acc[t] = wmma_bf16(a, bf, acc[t]);
    }
  }
  __bf16* outp = Obf + ((size_t)b * SEQ + q0) * DIM;
#pragma unroll
  for (int t = 0; t < 8; ++t) store_c_bf16(outp + n0 + 16 * t, DIM, acc[t]);
}

// ---------------------------------------------------------------------------
// Launch: q[B,S,D]f32, Wq[DK,D]f32, Wk[DK,D]f32, Wv[D,D]f32 -> out[B,S,D]f32
// Workspace (bf16): Qbf 16MB | Wqb .5MB | Wkb .5MB | Wvb 2MB | QP 4MB | KP 4MB
//                   | Obf 16MB | Vt 16MB   (~59 MB total)
// ---------------------------------------------------------------------------
extern "C" void kernel_launch(void* const* d_in, const int* in_sizes, int n_in,
                              void* d_out, int out_size, void* d_ws, size_t ws_size,
                              hipStream_t stream) {
  (void)in_sizes; (void)n_in; (void)out_size; (void)ws_size;
  const float* q  = (const float*)d_in[0];
  const float* Wq = (const float*)d_in[1];
  const float* Wk = (const float*)d_in[2];
  const float* Wv = (const float*)d_in[3];
  float* out = (float*)d_out;

  __bf16* Qbf = (__bf16*)d_ws;
  __bf16* Wqb = Qbf + (size_t)BATCH * SEQ * DIM;
  __bf16* Wkb = Wqb + (size_t)DKK * DIM;
  __bf16* Wvb = Wkb + (size_t)DKK * DIM;
  __bf16* QPp = Wvb + (size_t)DIM * DIM;
  __bf16* KPp = QPp + (size_t)BATCH * SEQ * DKK;
  __bf16* Obf = KPp + (size_t)BATCH * SEQ * DKK;
  __bf16* Vt  = Obf + (size_t)BATCH * SEQ * DIM;

  // 1) fp32 -> bf16 (+ transposed V copy)
  {
    int nq = BATCH * SEQ * DIM;
    cvt_f32_to_bf16<<<(nq + 255) / 256, 256, 0, stream>>>(q, Qbf, nq);
    int nw = DKK * DIM;
    cvt_f32_to_bf16<<<(nw + 255) / 256, 256, 0, stream>>>(Wq, Wqb, nw);
    cvt_f32_to_bf16<<<(nw + 255) / 256, 256, 0, stream>>>(Wk, Wkb, nw);
    int nv = DIM * DIM;
    cvt_f32_to_bf16<<<(nv + 255) / 256, 256, 0, stream>>>(Wv, Wvb, nv);
    dim3 tg(SEQ / 32, DIM / 32, BATCH);
    transpose_cvt_bf16<<<tg, dim3(32, 8), 0, stream>>>(q, Vt);
  }

  const size_t ldsA = (size_t)16 * DIM * sizeof(__bf16);  // 32 KB A tile

  // 2) QP = Qbf @ Wq^T, KP = Qbf @ Wk^T   (M=8192, N=256, K=1024), bf16 out
  {
    dim3 grid(DKK / 256, (BATCH * SEQ) / 16);
    gemm_a_bt<true><<<grid, 128, ldsA, stream>>>(Qbf, Wqb, QPp, BATCH * SEQ, DKK, DIM);
    gemm_a_bt<true><<<grid, 128, ldsA, stream>>>(Qbf, Wkb, KPp, BATCH * SEQ, DKK, DIM);
  }

  // 3) fused scores + softmax + P@V  (grid: 128 q-tiles x 4 batches)
  {
    dim3 grid(SEQ / 16, BATCH);
    size_t lds = (size_t)16 * SEQ * sizeof(float);  // 128 KB
    attn_softmax_pv<<<grid, 256, lds, stream>>>(QPp, KPp, Vt, Obf);
  }

  // 4) out = Obf @ Wv^T  (M=8192, N=1024, K=1024), fp32 out
  {
    dim3 grid(DIM / 256, (BATCH * SEQ) / 16);
    gemm_a_bt<false><<<grid, 128, ldsA, stream>>>(Obf, Wvb, out, BATCH * SEQ, DIM, DIM);
  }
}
